// LocalResponseNorm_81183471829681
// MI455X (gfx1250) — compile-verified, matrix operand order checked by reading, not run on hardware
//
#include <hip/hip_runtime.h>
#include <hip/hip_bf16.h>

// Problem constants (match reference setup_inputs)
#define BATCH 16
#define CHN   96
#define HDIM  224
#define WDIM  224
#define HW    (HDIM * WDIM)        // 50176, divisible by 32 and 4
#define KCONST 2.0f
#define ALPHA  1e-4f
#define BETA   0.75f

typedef __attribute__((ext_vector_type(2)))  float    v2f;
typedef __attribute__((ext_vector_type(8)))  float    v8f;
typedef __attribute__((ext_vector_type(16))) _Float16 v16h;

// ---------------------------------------------------------------------------
// Pass 1: s[b,h,w] = sum_c x[b,c,h,w]^2  via WMMA (A = ones, B = x^2 tile).
// Each wave handles 32 pixels (two groups of 16, two independent accumulators).
// With A == ones, D[m,n] = sum_k B[k,n]: any K-permutation of B's VGPR layout
// yields the same result, so we only need each (VGPR, lane-half) slot to carry
// a distinct channel of the 4-channel group.
// ---------------------------------------------------------------------------
__global__ __launch_bounds__(256) void lrn_chansum_wmma(const float* __restrict__ x,
                                                        float* __restrict__ s) {
    const int lane  = threadIdx.x & 31;
    const int half  = lane >> 4;       // 0: lanes 0-15, 1: lanes 16-31
    const int n     = lane & 15;       // pixel within 16-group
    const int gwave = blockIdx.x * 8 + (threadIdx.x >> 5);   // 8 waves / block

    const int  p0  = gwave * 32;                 // first pixel of this wave
    const int  b   = p0 / HW;                    // HW % 32 == 0 -> uniform b
    const int  hw0 = p0 - b * HW;
    const float* xb = x + (long)b * CHN * HW;

#if __has_builtin(__builtin_amdgcn_wmma_f32_16x16x4_f32)
    v2f a; a[0] = 1.0f; a[1] = 1.0f;             // A = ones(16x4)
    v8f c0 = {}; v8f c1 = {};
    for (int cb = 0; cb < CHN; cb += 4) {
        // pixel group A (p0 .. p0+15)
        {
            const int hwp = hw0 + n;
            float vx = xb[(long)(cb + half)     * HW + hwp];
            float vy = xb[(long)(cb + 2 + half) * HW + hwp];
            v2f bm; bm[0] = vx * vx; bm[1] = vy * vy;
            c0 = __builtin_amdgcn_wmma_f32_16x16x4_f32(false, a, false, bm,
                                                       (short)0, c0, false, false);
        }
        // pixel group B (p0+16 .. p0+31) -- independent accumulator for ILP
        {
            const int hwp = hw0 + 16 + n;
            float vx = xb[(long)(cb + half)     * HW + hwp];
            float vy = xb[(long)(cb + 2 + half) * HW + hwp];
            v2f bm; bm[0] = vx * vx; bm[1] = vy * vy;
            c1 = __builtin_amdgcn_wmma_f32_16x16x4_f32(false, a, false, bm,
                                                       (short)0, c1, false, false);
        }
    }
    // D row M=0 lives in accumulator VGPR0, lanes 0-15 (N = lane).
    if (lane < 16) {
        s[p0 + lane]      = c0[0];
        s[p0 + 16 + lane] = c1[0];
    }
#else
    // Fallback: confirmed f16 WMMA (x^2 well inside f16 range; C/D stay f32).
    v16h a;
#pragma unroll
    for (int i = 0; i < 16; ++i) a[i] = (_Float16)1.0f;
    v8f c0 = {}; v8f c1 = {};
    for (int cb = 0; cb < CHN; cb += 32) {       // 32 channels per WMMA, 3 iters
        v16h b0, b1;
#pragma unroll
        for (int h = 0; h < 16; ++h) {
            const int ch = cb + half * 16 + h;   // bijective slot->channel map
            float v0 = xb[(long)ch * HW + hw0 + n];
            float v1 = xb[(long)ch * HW + hw0 + 16 + n];
            b0[h] = (_Float16)(v0 * v0);
            b1[h] = (_Float16)(v1 * v1);
        }
        c0 = __builtin_amdgcn_wmma_f32_16x16x32_f16(false, a, false, b0,
                                                    (short)0, c0, false, false);
        c1 = __builtin_amdgcn_wmma_f32_16x16x32_f16(false, a, false, b1,
                                                    (short)0, c1, false, false);
    }
    if (lane < 16) {
        s[p0 + lane]      = c0[0];
        s[p0 + 16 + lane] = c1[0];
    }
#endif
}

// ---------------------------------------------------------------------------
// Pass 2: y = 5x5 zero-padded box filter of s; scale = (K + alpha*y)^(-beta).
// 800K pixels, all traffic L2-resident -- negligible cost.
// ---------------------------------------------------------------------------
__global__ __launch_bounds__(256) void lrn_boxpow(const float* __restrict__ s,
                                                  float* __restrict__ scale) {
    const int idx = blockIdx.x * 256 + threadIdx.x;     // exactly B*HW threads
    const int b  = idx / HW;
    const int hw = idx - b * HW;
    const int h  = hw / WDIM;
    const int w  = hw - h * WDIM;
    const float* sb = s + (long)b * HW;

    float acc = 0.0f;
#pragma unroll
    for (int dy = -2; dy <= 2; ++dy) {
        const int yy = h + dy;
        if (yy < 0 || yy >= HDIM) continue;
        const float* row = sb + yy * WDIM;
#pragma unroll
        for (int dx = -2; dx <= 2; ++dx) {
            const int xx = w + dx;
            if (xx < 0 || xx >= WDIM) continue;   // zero padding
            acc += row[xx];
        }
    }
    const float t = KCONST + ALPHA * acc;
    scale[idx] = powf(t, -BETA);                  // t > 0 always
}

// ---------------------------------------------------------------------------
// Pass 3: out = x * scale (broadcast over channels). float4 vectorized;
// scale is 3.2MB -> L2-resident with 96x reuse.
// ---------------------------------------------------------------------------
__global__ __launch_bounds__(256) void lrn_apply(const float* __restrict__ x,
                                                 const float* __restrict__ scale,
                                                 float* __restrict__ out) {
    const long base = ((long)blockIdx.x * 256 + threadIdx.x) * 4;
    const long chw  = (long)CHN * HW;
    const int  b    = (int)(base / chw);
    const int  hw   = (int)((base - (long)b * chw) % HW);   // 4 elems stay in-row

    const float4 xv = *reinterpret_cast<const float4*>(x + base);
    const float4 sv = *reinterpret_cast<const float4*>(scale + (long)b * HW + hw);
    float4 o;
    o.x = xv.x * sv.x;
    o.y = xv.y * sv.y;
    o.z = xv.z * sv.z;
    o.w = xv.w * sv.w;
    *reinterpret_cast<float4*>(out + base) = o;
}

extern "C" void kernel_launch(void* const* d_in, const int* in_sizes, int n_in,
                              void* d_out, int out_size, void* d_ws, size_t ws_size,
                              hipStream_t stream) {
    const float* x   = (const float*)d_in[0];
    float*       out = (float*)d_out;

    // Scratch: s [B*HW] then scale [B*HW]  (needs 2 * 3.2 MB)
    float* s     = (float*)d_ws;
    float* scale = s + (size_t)BATCH * HW;

    const dim3 blk(256);

    // Pass 1: 25088 waves (32 px each) = 3136 blocks of 8 waves.
    const int waves1 = (BATCH * HW) / 32;
    lrn_chansum_wmma<<<dim3(waves1 / 8), blk, 0, stream>>>(x, s);

    // Pass 2: one thread per pixel map element (exactly divisible).
    lrn_boxpow<<<dim3((BATCH * HW) / 256), blk, 0, stream>>>(s, scale);

    // Pass 3: float4 per thread over the full tensor (exactly divisible).
    const long total = (long)BATCH * CHN * HW;
    lrn_apply<<<dim3((unsigned)(total / 4 / 256)), blk, 0, stream>>>(x, scale, out);
}
